// PIRAModel_65146063945948
// MI455X (gfx1250) — matrix operand order repaired and checked by reading, not compiled, not run on hardware
//
#include <hip/hip_runtime.h>
#include <stdint.h>

#define NN      50000
#define EE      1600000
#define ETOT    (EE + NN)     // edges + self loops
#define HID     128
#define HEADS   4
#define HD      32
#define FIN     39
#define LN_EPS  1e-5f

typedef __attribute__((ext_vector_type(16))) _Float16 v16h;
typedef __attribute__((ext_vector_type(8)))  float    v8f;

union Frag16 { v16h h; uint32_t u[8]; };

// ---------------- helpers ----------------

// monotonic float<->uint encoding for atomic max on floats (handles negatives)
__device__ __forceinline__ unsigned enc_f(float v) {
  unsigned b = __float_as_uint(v);
  return (b & 0x80000000u) ? ~b : (b | 0x80000000u);
}
__device__ __forceinline__ float dec_f(unsigned u) {
  return __uint_as_float((u & 0x80000000u) ? (u & 0x7fffffffu) : ~u);
}

// A-fragment K offset within a 32-wide K step (ISA 7.12.2, 16-bit A 16x32):
// VGPR 0..3: K = hsel*8 + 2p (+1), VGPR 4..7: K = 16 + hsel*8 + 2(p-4)
__device__ __forceinline__ int a_koff(int p, int hsel) {
  return ((p & 4) << 2) + hsel * 8 + ((p & 3) << 1);
}
// B-fragment K offset (same K order as A):
// VGPR 0..3: K = hsel*8 + 2p (+1), VGPR 4..7: K = 16 + hsel*8 + 2(p-4)
__device__ __forceinline__ int b_koff(int p, int hsel) {
  return ((p & 4) << 2) + hsel * 8 + ((p & 3) << 1);
}

// ---------------- weight prep: fp32 [K][M] -> f16 transposed [M][K] ----------------
__global__ __launch_bounds__(256)
void wconv_kernel(const float* __restrict__ W, _Float16* __restrict__ Bt) {
  int idx = blockIdx.x * 256 + threadIdx.x;     // 128*128
  int m = idx >> 7, k = idx & 127;
  Bt[m * 128 + k] = (_Float16)W[k * 128 + m];
}

// ---------------- encoder layer 1: t16 = relu(x @ enc_w1 + b1), f16 out ----------------
__global__ __launch_bounds__(128)
void enc1_kernel(const float* __restrict__ x, const float* __restrict__ w1,
                 const float* __restrict__ b1, _Float16* __restrict__ t16) {
  __shared__ float xs[FIN];
  const long n = blockIdx.x;
  const int  m = threadIdx.x;
  if (m < FIN) xs[m] = x[n * FIN + m];
  __syncthreads();
  float acc = b1[m];
  #pragma unroll
  for (int k = 0; k < FIN; ++k) acc += xs[k] * w1[k * HID + m];
  t16[n * HID + m] = (_Float16)fmaxf(acc, 0.0f);
}

// ---------------- WMMA GEMM: C[N x 128] = A16[N x 128] @ B (+bias) ----------------
// Bt16 is the weight pre-transposed to [n][k] f16 so K-pairs are dword-contiguous.
// A block-tile (128 rows x 128 K f16) is staged into LDS with the gfx1250 async
// memory->LDS path (global_load_async_to_lds_b128, ASYNCcnt), then 8 waves each
// own one 16-wide column tile and sweep 8 row tiles x 4 K-steps of
// v_wmma_f32_16x16x32_f16.
#define LDA 68   // dwords per LDS row: 272 B = 17*16 B -> b128-aligned, banks shift by 4/row

template<int DO_RELU, int WRITE16, int HAS_BIAS>
__global__ __launch_bounds__(256)
void gemm128_wmma(const _Float16* __restrict__ A16, const _Float16* __restrict__ Bt16,
                  const float* __restrict__ bias, float* __restrict__ C,
                  _Float16* __restrict__ C16, int nrows) {
  __shared__ uint32_t ldsA[128 * LDA];
  const int t      = threadIdx.x;
  const int wave   = t >> 5;
  const int lane   = t & 31;
  const int lane15 = lane & 15;
  const int hsel   = lane >> 4;
  const long rowbase = (long)blockIdx.x * 128;

  // ---- async stage A block-tile: 2048 x b128 chunks, 8 per thread ----
  const uint32_t* A32 = (const uint32_t*)A16;
  #pragma unroll
  for (int it = 0; it < 8; ++it) {
    int idx = it * 256 + t;          // 0..2047
    int r = idx >> 4;                // row in tile (16 b128 per 128-half row)
    int d = idx & 15;                // which 16B chunk
    long row = rowbase + r;
    if (row >= nrows) row = nrows - 1;
    unsigned ldsaddr = (unsigned)(uintptr_t)&ldsA[r * LDA + d * 4];
    int      goff    = (int)((row * 64 + d * 4) * 4);   // byte offset from A32
    asm volatile("global_load_async_to_lds_b128 %0, %1, %2"
                 :: "v"(ldsaddr), "v"(goff), "s"(A32) : "memory");
  }
  asm volatile("s_wait_asynccnt 0x0" ::: "memory");
  __syncthreads();

  // ---- this wave's column tile; preload 4 B fragments (weights are L2-resident) ----
  const int cb = wave * 16;
  const int n  = cb + lane15;
  const uint32_t* B32 = (const uint32_t*)Bt16;
  Frag16 bf[4];
  #pragma unroll
  for (int ks = 0; ks < 4; ++ks)
    #pragma unroll
    for (int p = 0; p < 8; ++p) {
      int k = ks * 32 + b_koff(p, hsel);
      bf[ks].u[p] = B32[n * 64 + (k >> 1)];
    }

  const float bn = HAS_BIAS ? bias[n] : 0.0f;
  const bool  full = (rowbase + 128 <= (long)nrows);

  #pragma unroll
  for (int rt = 0; rt < 8; ++rt) {
    v8f acc = {};
    #pragma unroll
    for (int ks = 0; ks < 4; ++ks) {
      Frag16 af;
      #pragma unroll
      for (int p = 0; p < 8; ++p) {
        int k = ks * 32 + a_koff(p, hsel);
        af.u[p] = ldsA[(rt * 16 + lane15) * LDA + (k >> 1)];
      }
      acc = __builtin_amdgcn_wmma_f32_16x16x32_f16(false, af.h, false, bf[ks].h,
                                                   (short)0, acc, false, false);
    }
    if (full) {
      #pragma unroll
      for (int r = 0; r < 8; ++r) {
        long m = rowbase + rt * 16 + hsel * 8 + r;
        float v = acc[r] + bn;
        if (DO_RELU) v = fmaxf(v, 0.0f);
        C[m * HID + n] = v;
        if (WRITE16) C16[m * HID + n] = (_Float16)v;
      }
    } else {
      #pragma unroll
      for (int r = 0; r < 8; ++r) {
        long m = rowbase + rt * 16 + hsel * 8 + r;
        if (m < nrows) {
          float v = acc[r] + bn;
          if (DO_RELU) v = fmaxf(v, 0.0f);
          C[m * HID + n] = v;
          if (WRITE16) C16[m * HID + n] = (_Float16)v;
        }
      }
    }
  }
}

// ---------------- attention scores per (node, head) ----------------
__global__ __launch_bounds__(256)
void att_scores_kernel(const float* __restrict__ xp, const float* __restrict__ a_s,
                       const float* __restrict__ a_d, float* __restrict__ es,
                       float* __restrict__ ed) {
  int idx = blockIdx.x * 256 + threadIdx.x;     // N*HEADS
  if (idx >= NN * HEADS) return;
  int h = idx & 3;
  const float* v = xp + (long)(idx >> 2) * HID + h * HD;
  float s = 0.f, d = 0.f;
  #pragma unroll 8
  for (int c = 0; c < HD; ++c) { s += v[c] * a_s[h * HD + c]; d += v[c] * a_d[h * HD + c]; }
  es[idx] = s; ed[idx] = d;
}

__device__ __forceinline__ void edge_sd(int e, const int* src, const int* dst, int& s, int& d) {
  if (e < EE) { s = src[e]; d = dst[e]; } else { s = e - EE; d = s; }
}

// ---------------- pass 1: segment max (atomic max via monotonic uint) ----------------
__global__ __launch_bounds__(256)
void edge_max_kernel(const int* __restrict__ src, const int* __restrict__ dst,
                     const float* __restrict__ es, const float* __restrict__ ed,
                     unsigned* __restrict__ emax_u) {
  int e = blockIdx.x * 256 + threadIdx.x;
  if (e >= ETOT) return;
  int s, d; edge_sd(e, src, dst, s, d);
  #pragma unroll
  for (int h = 0; h < HEADS; ++h) {
    float v = es[s * HEADS + h] + ed[d * HEADS + h];
    v = v > 0.f ? v : 0.2f * v;                  // leaky_relu(0.2)
    atomicMax(&emax_u[d * HEADS + h], enc_f(v));
  }
}

// ---------------- pass 2: softmax denominator ----------------
__global__ __launch_bounds__(256)
void edge_denom_kernel(const int* __restrict__ src, const int* __restrict__ dst,
                       const float* __restrict__ es, const float* __restrict__ ed,
                       const unsigned* __restrict__ emax_u, float* __restrict__ denom) {
  int e = blockIdx.x * 256 + threadIdx.x;
  if (e >= ETOT) return;
  int s, d; edge_sd(e, src, dst, s, d);
  #pragma unroll
  for (int h = 0; h < HEADS; ++h) {
    float v = es[s * HEADS + h] + ed[d * HEADS + h];
    v = v > 0.f ? v : 0.2f * v;
    float a = __expf(v - dec_f(emax_u[d * HEADS + h]));
    __hip_atomic_fetch_add(&denom[d * HEADS + h], a, __ATOMIC_RELAXED, __HIP_MEMORY_SCOPE_AGENT);
  }
}

// ---------------- pass 3: weighted scatter-add, one wave per edge ----------------
__global__ __launch_bounds__(256)
void edge_scatter_kernel(const int* __restrict__ src, const int* __restrict__ dst,
                         const float* __restrict__ es, const float* __restrict__ ed,
                         const unsigned* __restrict__ emax_u, const float* __restrict__ denom,
                         const float* __restrict__ xp, float* __restrict__ gat) {
  long gid = (long)blockIdx.x * 256 + threadIdx.x;
  int  e   = (int)(gid >> 5);
  int  lane = (int)(gid & 31);
  if (e >= ETOT) return;
  int s, d; edge_sd(e, src, dst, s, d);
  float w[HEADS];
  #pragma unroll
  for (int h = 0; h < HEADS; ++h) {
    float v = es[s * HEADS + h] + ed[d * HEADS + h];
    v = v > 0.f ? v : 0.2f * v;
    w[h] = __expf(v - dec_f(emax_u[d * HEADS + h])) / denom[d * HEADS + h];
  }
  #pragma unroll
  for (int q = 0; q < 4; ++q) {
    int j = lane + q * 32;                        // channel; head = q
    float val = xp[(long)s * HID + j] * w[q];
    __hip_atomic_fetch_add(&gat[(long)d * HID + j], val,
                           __ATOMIC_RELAXED, __HIP_MEMORY_SCOPE_AGENT);
  }
}

// ---------------- residual + layernorm + relu; one wave per node ----------------
__global__ __launch_bounds__(256)
void ln_kernel(const float* __restrict__ gat, const float* __restrict__ bc,
               const float* __restrict__ g, const float* __restrict__ be,
               float* __restrict__ h, _Float16* __restrict__ h16) {
  int node = blockIdx.x * 8 + (threadIdx.x >> 5);
  int lane = threadIdx.x & 31;
  if (node >= NN) return;
  float v[4]; float sum = 0.f;
  #pragma unroll
  for (int q = 0; q < 4; ++q) {
    int j = lane + q * 32;
    v[q] = gat[(long)node * HID + j] + bc[j] + h[(long)node * HID + j];
    sum += v[q];
  }
  #pragma unroll
  for (int off = 16; off; off >>= 1) sum += __shfl_xor(sum, off, 32);
  float mean = sum * (1.0f / HID);
  float var = 0.f;
  #pragma unroll
  for (int q = 0; q < 4; ++q) { float t = v[q] - mean; var += t * t; }
  #pragma unroll
  for (int off = 16; off; off >>= 1) var += __shfl_xor(var, off, 32);
  float inv = rsqrtf(var * (1.0f / HID) + LN_EPS);
  #pragma unroll
  for (int q = 0; q < 4; ++q) {
    int j = lane + q * 32;
    float y = fmaxf((v[q] - mean) * inv * g[j] + be[j], 0.0f);
    h[(long)node * HID + j]   = y;
    h16[(long)node * HID + j] = (_Float16)y;
  }
}

// ---------------- output heads: one node per 64-thread block ----------------
__global__ __launch_bounds__(64)
void head_kernel(const float* __restrict__ h, const float* __restrict__ w1,
                 const float* __restrict__ b1, const float* __restrict__ w2,
                 const float* __restrict__ b2, float* __restrict__ out,
                 int outd, int use_softplus) {
  __shared__ float hs[HID];
  __shared__ float hid[64];
  const long node = blockIdx.x;
  const int  t = threadIdx.x;
  hs[t]      = h[node * HID + t];
  hs[t + 64] = h[node * HID + t + 64];
  __syncthreads();
  float acc = b1[t];
  #pragma unroll 8
  for (int k = 0; k < HID; ++k) acc += hs[k] * w1[k * 64 + t];
  hid[t] = fmaxf(acc, 0.0f);
  __syncthreads();
  if (t < outd) {
    float o = b2[t];
    #pragma unroll 8
    for (int k = 0; k < 64; ++k) o += hid[k] * w2[k * outd + t];
    if (use_softplus) o = (o > 20.0f) ? o : log1pf(__expf(o));
    out[node * outd + t] = o;
  }
}

// ---------------- launcher ----------------
extern "C" void kernel_launch(void* const* d_in, const int* in_sizes, int n_in,
                              void* d_out, int out_size, void* d_ws, size_t ws_size,
                              hipStream_t stream) {
  const float* x      = (const float*)d_in[0];
  const int*   ei     = (const int*)d_in[1];
  const int*   src    = ei;
  const int*   dst    = ei + EE;
  const float* enc_w1 = (const float*)d_in[2];
  const float* enc_b1 = (const float*)d_in[3];
  const float* enc_w2 = (const float*)d_in[4];
  const float* enc_b2 = (const float*)d_in[5];
  const float *Wl[3], *asl[3], *adl[3], *bcl[3], *gl[3], *bel[3];
  for (int l = 0; l < 3; ++l) {
    int b = 6 + l * 6;
    Wl[l]  = (const float*)d_in[b + 0];
    asl[l] = (const float*)d_in[b + 1];
    adl[l] = (const float*)d_in[b + 2];
    bcl[l] = (const float*)d_in[b + 3];
    gl[l]  = (const float*)d_in[b + 4];
    bel[l] = (const float*)d_in[b + 5];
  }
  const float* imp_w1 = (const float*)d_in[24];
  const float* imp_b1 = (const float*)d_in[25];
  const float* imp_w2 = (const float*)d_in[26];
  const float* imp_b2 = (const float*)d_in[27];
  const float* tim_w1 = (const float*)d_in[28];
  const float* tim_b1 = (const float*)d_in[29];
  const float* tim_w2 = (const float*)d_in[30];
  const float* tim_b2 = (const float*)d_in[31];
  float* out = (float*)d_out;

  // carve workspace
  char* base = (char*)d_ws;
  size_t off = 0;
  auto carve = [&](size_t bytes) -> char* {
    char* p = base + off;
    off = (off + bytes + 255) & ~(size_t)255;
    return p;
  };
  float*     h      = (float*)carve((size_t)NN * HID * 4);
  _Float16*  h16    = (_Float16*)carve((size_t)NN * HID * 2);
  float*     xp     = (float*)carve((size_t)NN * HID * 4);
  float*     gat    = (float*)carve((size_t)NN * HID * 4);
  _Float16*  t16    = (_Float16*)carve((size_t)NN * HID * 2);
  float*     es     = (float*)carve((size_t)NN * HEADS * 4);
  float*     ed     = (float*)carve((size_t)NN * HEADS * 4);
  unsigned*  emax_u = (unsigned*)carve((size_t)NN * HEADS * 4);
  float*     denom  = (float*)carve((size_t)NN * HEADS * 4);
  _Float16*  wt[4];
  for (int i = 0; i < 4; ++i) wt[i] = (_Float16*)carve((size_t)HID * HID * 2);

  // 0) pre-convert the four 128x128 weight matrices to f16, transposed [n][k]
  wconv_kernel<<<64, 256, 0, stream>>>(enc_w2, wt[0]);
  for (int l = 0; l < 3; ++l) wconv_kernel<<<64, 256, 0, stream>>>(Wl[l], wt[l + 1]);

  // 1) encoder
  enc1_kernel<<<NN, 128, 0, stream>>>(x, enc_w1, enc_b1, t16);
  int gblocks = (NN + 127) / 128;
  gemm128_wmma<0, 1, 1><<<gblocks, 256, 0, stream>>>(t16, wt[0], enc_b2, h, h16, NN);

  // 2) three GAT layers
  const int eb  = (ETOT + 255) / 256;
  const int sb  = (int)(((long)ETOT * 32 + 255) / 256);
  const int ab  = (NN * HEADS + 255) / 256;
  const int lnb = (NN + 7) / 8;
  for (int l = 0; l < 3; ++l) {
    gemm128_wmma<0, 0, 0><<<gblocks, 256, 0, stream>>>(h16, wt[l + 1], nullptr, xp, nullptr, NN);
    att_scores_kernel<<<ab, 256, 0, stream>>>(xp, asl[l], adl[l], es, ed);
    hipMemsetAsync(emax_u, 0, (size_t)NN * HEADS * 4, stream);   // enc(min)
    hipMemsetAsync(denom,  0, (size_t)NN * HEADS * 4, stream);
    hipMemsetAsync(gat,    0, (size_t)NN * HID * 4, stream);
    edge_max_kernel<<<eb, 256, 0, stream>>>(src, dst, es, ed, emax_u);
    edge_denom_kernel<<<eb, 256, 0, stream>>>(src, dst, es, ed, emax_u, denom);
    edge_scatter_kernel<<<sb, 256, 0, stream>>>(src, dst, es, ed, emax_u, denom, xp, gat);
    ln_kernel<<<lnb, 256, 0, stream>>>(gat, bcl[l], gl[l], bel[l], h, h16);
  }

  // 3) heads: impact (N*3) then timing (N*2), concatenated
  head_kernel<<<NN, 64, 0, stream>>>(h, imp_w1, imp_b1, imp_w2, imp_b2, out, 3, 0);
  head_kernel<<<NN, 64, 0, stream>>>(h, tim_w1, tim_b1, tim_w2, tim_b2, out + (size_t)NN * 3, 2, 1);
}
